// ScannedRNN_36859409334883
// MI455X (gfx1250) — compile-verified, matrix operand order checked by reading, not run on hardware
//
#include <hip/hip_runtime.h>
#include <stdint.h>
#include <stddef.h>

// Problem dims (fixed by the reference): T=256, B=64, A=8, H=512
#define TSTEPS  256
#define ROWS    512                 // B*A sequences
#define HDIM    512
#define G3      1536                // 3*H
#define TOTROWS (TSTEPS * ROWS)     // 131072

// gru_scan LDS layout (bytes)
#define LDS_B_BYTES 98304           // 6 tiles (2 coltiles x 3 gates) x 16 x 512 bf16
#define LDS_A_OFF   98304
#define LDS_A_BYTES 65536           // 64 rows x 512 bf16 slab of h
#define LDS_TOTAL   (LDS_B_BYTES + LDS_A_BYTES)   // 160 KB (<320 KB/WGP)

typedef __attribute__((ext_vector_type(16))) __bf16 v16bf;
typedef __attribute__((ext_vector_type(8)))  float  v8f;
typedef unsigned int u32x4 __attribute__((ext_vector_type(4)));
typedef float        f32x4 __attribute__((ext_vector_type(4)));

union V16 {
    v16bf v;
    u32x4 q[2];
};

static __device__ __forceinline__ v8f wmma_bf16(const V16& a, const V16& b, v8f c) {
    // (neg_a, A, neg_b, B, c_mod, C, reuse_a, reuse_b)
    return __builtin_amdgcn_wmma_f32_16x16x32_bf16(false, a.v, false, b.v, (short)0, c,
                                                   false, false);
}

// CDNA5 async copy: global -> LDS, tracked by ASYNCcnt (CDNA5 ISA)
static __device__ __forceinline__ void async_g2l_b128(unsigned lds_byte_off,
                                                      const void* gaddr) {
    asm volatile("global_load_async_to_lds_b128 %0, %1, off"
                 :: "v"(lds_byte_off), "v"(gaddr)
                 : "memory");
}
static __device__ __forceinline__ void wait_asynccnt0() {
    asm volatile("s_wait_asynccnt 0x0" ::: "memory");
}

// ---------------------------------------------------------------------------
// Prep: transpose Wi/Wh [H,3H] fp32 -> [3H,H] bf16, zero h buffers, reset barrier
// ---------------------------------------------------------------------------
__global__ void gru_prep(const float* __restrict__ Wi, const float* __restrict__ Wh,
                         __bf16* __restrict__ WiT, __bf16* __restrict__ WhT,
                         float* __restrict__ h0, float* __restrict__ h1,
                         __bf16* __restrict__ hb0, __bf16* __restrict__ hb1,
                         unsigned* __restrict__ bar) {
    const int idx    = blockIdx.x * blockDim.x + threadIdx.x;
    const int stride = gridDim.x * blockDim.x;
    for (int e = idx; e < G3 * HDIM; e += stride) {
        const int g = e / HDIM;
        const int k = e - g * HDIM;
        WiT[e] = (__bf16)Wi[(size_t)k * G3 + g];
        WhT[e] = (__bf16)Wh[(size_t)k * G3 + g];
    }
    for (int e = idx; e < ROWS * HDIM; e += stride) {
        h0[e] = 0.f;
        h1[e] = 0.f;
        hb0[e] = (__bf16)0.f;
        hb1[e] = (__bf16)0.f;
    }
    if (idx == 0) *bar = 0u;
}

// ---------------------------------------------------------------------------
// Input-side GEMM: XG[TOTROWS,G3] = bf16(X[TOTROWS,H]) @ bf16(Wi) + bi (fp32 acc)
// One 16x16 output tile per wave, grid-stride over tiles. Memory-bound (~1 GB).
// ---------------------------------------------------------------------------
__global__ void __launch_bounds__(256)
xgates_gemm(const float* __restrict__ X,      // [TOTROWS, H] fp32
            const __bf16* __restrict__ WiT,   // [G3, H] bf16 (transposed)
            const float* __restrict__ bi,     // [G3]
            float* __restrict__ XG) {         // [TOTROWS, G3]
    const int lane   = threadIdx.x & 31;
    const int wave   = (blockIdx.x * blockDim.x + threadIdx.x) >> 5;
    const int nwaves = (gridDim.x * blockDim.x) >> 5;
    const int half   = lane >> 4;     // K-chunk select per WMMA 16-bit layout
    const int l16    = lane & 15;
    const int rtiles = TOTROWS / 16;  // 8192
    const int ctiles = G3 / 16;       // 96
    const int ntiles = rtiles * ctiles;

    for (int tile = wave; tile < ntiles; tile += nwaves) {
        const int rt = tile / ctiles;
        const int ct = tile - rt * ctiles;
        const int arow = rt * 16 + l16;   // A: M = lane&15
        const int brow = ct * 16 + l16;   // B: N = lane&15 (WiT row = output col)
        v8f acc = {};
        for (int k0 = 0; k0 < HDIM; k0 += 32) {
            // A operand: 16 fp32 -> 16 bf16 (chunks at k0+half*8 and +16)
            const float* ap = X + (size_t)arow * HDIM + k0 + half * 8;
            f32x4 f0 = *(const f32x4*)(ap);
            f32x4 f1 = *(const f32x4*)(ap + 4);
            f32x4 f2 = *(const f32x4*)(ap + 16);
            f32x4 f3 = *(const f32x4*)(ap + 20);
            V16 a;
#pragma unroll
            for (int e = 0; e < 4; ++e) {
                a.v[e]      = (__bf16)f0[e];
                a.v[4 + e]  = (__bf16)f1[e];
                a.v[8 + e]  = (__bf16)f2[e];
                a.v[12 + e] = (__bf16)f3[e];
            }
            // B operand: two b128 loads of bf16
            const __bf16* bp = WiT + (size_t)brow * HDIM + k0 + half * 8;
            V16 b;
            b.q[0] = *(const u32x4*)(bp);
            b.q[1] = *(const u32x4*)(bp + 16);
            acc = wmma_bf16(a, b, acc);
        }
        // C/D layout: VGPR v -> M = v + half*8 ; lane&15 -> N
#pragma unroll
        for (int v = 0; v < 8; ++v) {
            const int row = rt * 16 + v + half * 8;
            const int col = ct * 16 + l16;
            XG[(size_t)row * G3 + col] = acc[v] + bi[col];
        }
    }
}

// ---------------------------------------------------------------------------
// Persistent GRU scan: 128 blocks x 256 threads = 1024 wave32s.
// Block b: row-group rg=b>>4 (4 row-tiles), col-group cg=b&15 (2 col-tiles).
// Wave w (0..7): ti = rg*4 + (w>>1), tj = cg*2 + (w&1).
// B tiles (step-invariant, 96 KB) preloaded to LDS once via async-to-LDS;
// A slab (64 rows of h_bf16, 64 KB) re-staged to LDS every step.
// ---------------------------------------------------------------------------
__global__ void __launch_bounds__(256, 1)
gru_scan(const float* __restrict__ XG,            // [TOTROWS, G3]
         const unsigned char* __restrict__ resets,// [TSTEPS*ROWS] bool
         const __bf16* __restrict__ WhT,          // [G3, H] bf16 (transposed)
         const float* __restrict__ bhn,           // [H]
         float* __restrict__ h0, float* __restrict__ h1,
         __bf16* __restrict__ hb0, __bf16* __restrict__ hb1,
         float* __restrict__ Y,                   // [TOTROWS, H]
         unsigned* __restrict__ bar) {
    extern __shared__ __align__(16) char smem[];  // LDS_TOTAL bytes

    const int tid  = threadIdx.x;
    const int lane = tid & 31;
    const int w    = tid >> 5;               // wave in block, 0..7
    const int rg   = blockIdx.x >> 4;        // 0..7  -> row tiles rg*4 .. rg*4+3
    const int cg   = blockIdx.x & 15;        // 0..15 -> col tiles cg*2, cg*2+1
    const int tjl  = w & 1;                  // local col tile
    const int half = lane >> 4;
    const int l16  = lane & 15;

    const int ti = rg * 4 + (w >> 1);        // global row tile 0..31
    const int tj = cg * 2 + tjl;             // global col tile 0..31

    // -------- one-time: preload 6 B tiles (2 coltiles x 3 gates) into LDS ----
    // LDS B layout: tile (tjl*3+g) -> [16 rows][512 k] bf16, 1 KB per row.
    for (int c = tid; c < LDS_B_BYTES / 16; c += 256) {
        const int e    = c * 8;              // bf16 element offset
        const int tile = e >> 13;            // / (16*512)
        const int rem  = e & 8191;
        const int row  = rem >> 9;
        const int k    = rem & 511;
        const int tl   = tile / 3;
        const int g    = tile - tl * 3;
        const __bf16* src =
            WhT + ((size_t)(g * HDIM + (cg * 2 + tl) * 16 + row)) * HDIM + k;
        async_g2l_b128((unsigned)(uintptr_t)(smem + c * 16), src);
    }
    wait_asynccnt0();
    __syncthreads();

    // per-wave LDS read bases (byte offsets handled inline)
    const int aRow  = (w >> 1) * 16 + l16;            // row within 64-row A slab
    const int bRowR = (tjl * 3 + 0) * 16 + l16;       // rows within B region
    const int bRowZ = (tjl * 3 + 1) * 16 + l16;
    const int bRowN = (tjl * 3 + 2) * 16 + l16;

    float*  hcur = h0;  float*  hnxt = h1;
    __bf16* hbc  = hb0; __bf16* hbn  = hb1;

    const unsigned nblocks = gridDim.x;

    for (int t = 0; t < TSTEPS; ++t) {
        // ---- stage A slab: rows [rg*64, rg*64+64) of h_bf16 -> LDS ----
        {
            const __bf16* slab = hbc + (size_t)rg * 64 * HDIM;
            for (int c = tid; c < LDS_A_BYTES / 16; c += 256) {
                async_g2l_b128((unsigned)(uintptr_t)(smem + LDS_A_OFF + c * 16),
                               slab + c * 8);
            }
            wait_asynccnt0();
            __syncthreads();
        }

        // ---- hg = h @ Wh for 3 gates, all operands from LDS ----
        v8f accR = {}, accZ = {}, accN = {};
        for (int k0 = 0; k0 < HDIM; k0 += 32) {
            const int kb = (k0 + half * 8) * 2;       // byte offset of chunk0
            V16 a, br, bz, bn;
            const char* ap = smem + LDS_A_OFF + aRow * 1024 + kb;
            a.q[0] = *(const u32x4*)(ap);
            a.q[1] = *(const u32x4*)(ap + 32);
            const char* pr = smem + bRowR * 1024 + kb;
            br.q[0] = *(const u32x4*)(pr);  br.q[1] = *(const u32x4*)(pr + 32);
            const char* pz = smem + bRowZ * 1024 + kb;
            bz.q[0] = *(const u32x4*)(pz);  bz.q[1] = *(const u32x4*)(pz + 32);
            const char* pn = smem + bRowN * 1024 + kb;
            bn.q[0] = *(const u32x4*)(pn);  bn.q[1] = *(const u32x4*)(pn + 32);
            accR = wmma_bf16(a, br, accR);
            accZ = wmma_bf16(a, bz, accZ);
            accN = wmma_bf16(a, bn, accN);
        }

        // ---- fused GRU gate math on the 8 C/D elements this lane owns ----
#pragma unroll
        for (int v = 0; v < 8; ++v) {
            const int row = ti * 16 + v + half * 8;
            const int col = tj * 16 + l16;
            const size_t grow = (size_t)t * ROWS + row;
            const float xr = XG[grow * G3 + col];
            const float xz = XG[grow * G3 + HDIM + col];
            const float xn = XG[grow * G3 + 2 * HDIM + col];
            const float hp = hcur[(size_t)row * HDIM + col];
            const float r  = 1.f / (1.f + __expf(-(xr + accR[v])));
            const float z  = 1.f / (1.f + __expf(-(xz + accZ[v])));
            const float n  = tanhf(xn + r * (accN[v] + bhn[col]));
            const float nh = (1.f - z) * n + z * hp;
            Y[grow * HDIM + col] = nh;
            const float carry = resets[grow] ? 0.f : nh;
            hnxt[(size_t)row * HDIM + col] = carry;
            hbn[(size_t)row * HDIM + col]  = (__bf16)carry;
        }

        // swap double buffers
        { float*  tp = hcur; hcur = hnxt; hnxt = tp; }
        { __bf16* tp = hbc;  hbc  = hbn;  hbn  = tp; }

        // device-wide barrier: all stores of step t visible before step t+1 reads
        __threadfence();
        __syncthreads();
        if (tid == 0) {
            atomicAdd(bar, 1u);
            const unsigned target = (unsigned)(t + 1) * nblocks;
            while (atomicAdd(bar, 0u) < target) {
                __builtin_amdgcn_s_sleep(2);
            }
        }
        __syncthreads();
        __threadfence();
    }
}

// ---------------------------------------------------------------------------
extern "C" void kernel_launch(void* const* d_in, const int* in_sizes, int n_in,
                              void* d_out, int out_size, void* d_ws, size_t ws_size,
                              hipStream_t stream) {
    const float*         ins    = (const float*)d_in[0];         // [T,B,A,H] fp32
    const unsigned char* resets = (const unsigned char*)d_in[1]; // [T,B,A] bool
    const float*         Wi     = (const float*)d_in[2];         // [H,3H]
    const float*         bi     = (const float*)d_in[3];         // [3H]
    const float*         Wh     = (const float*)d_in[4];         // [H,3H]
    const float*         bhn    = (const float*)d_in[5];         // [H]
    float*               Y      = (float*)d_out;                 // [T,B,A,H]

    char* ws = (char*)d_ws;
    size_t off = 0;
    float*  XG  = (float*)(ws + off);  off += (size_t)TOTROWS * G3 * sizeof(float);
    __bf16* WiT = (__bf16*)(ws + off); off += (size_t)G3 * HDIM * sizeof(__bf16);
    __bf16* WhT = (__bf16*)(ws + off); off += (size_t)G3 * HDIM * sizeof(__bf16);
    float*  h0  = (float*)(ws + off);  off += (size_t)ROWS * HDIM * sizeof(float);
    float*  h1  = (float*)(ws + off);  off += (size_t)ROWS * HDIM * sizeof(float);
    __bf16* hb0 = (__bf16*)(ws + off); off += (size_t)ROWS * HDIM * sizeof(__bf16);
    __bf16* hb1 = (__bf16*)(ws + off); off += (size_t)ROWS * HDIM * sizeof(__bf16);
    unsigned* bar = (unsigned*)(ws + off);

    // allow 160 KB dynamic LDS (CDNA5 WGP has 320 KB)
    (void)hipFuncSetAttribute((const void*)gru_scan,
                              hipFuncAttributeMaxDynamicSharedMemorySize, LDS_TOTAL);

    gru_prep<<<256, 256, 0, stream>>>(Wi, Wh, WiT, WhT, h0, h1, hb0, hb1, bar);
    xgates_gemm<<<4096, 256, 0, stream>>>(ins, WiT, bi, XG);
    gru_scan<<<128, 256, LDS_TOTAL, stream>>>(XG, resets, WhT, bhn,
                                              h0, h1, hb0, hb1, Y, bar);
}
